// GCNModule_32126355374906
// MI455X (gfx1250) — matrix operand order repaired and checked by reading, not compiled, not run on hardware
//
#include <hip/hip_runtime.h>

typedef __attribute__((ext_vector_type(16))) _Float16 v16h;
typedef __attribute__((ext_vector_type(8)))  _Float16 v8h;
typedef __attribute__((ext_vector_type(8)))  float    v8f;

// ---------------- small utility kernels ----------------

__global__ void zero_i32_kernel(int* __restrict__ p, int n) {
    int i = blockIdx.x * blockDim.x + threadIdx.x;
    if (i < n) p[i] = 0;
}

// deg[m] = #edges with src==m   (row sums of A minus the self loop)
__global__ void degree_kernel(const int* __restrict__ ei, int E, int* __restrict__ deg) {
    int e = blockIdx.x * blockDim.x + threadIdx.x;
    if (e < E) atomicAdd(&deg[ei[e]], 1);
}

__global__ void dinv_kernel(const int* __restrict__ deg, float* __restrict__ dinv, int n) {
    int i = blockIdx.x * blockDim.x + threadIdx.x;
    if (i < n) dinv[i] = rsqrtf((float)deg[i] + 1.0f);   // +1 self loop; always > 0
}

// LDS-tiled transpose: in [B, R, Cc] -> out [B, Cc, R]
__global__ void transpose_bmn_kernel(const float* __restrict__ in, float* __restrict__ out,
                                     int R, int Cc) {
    __shared__ float tile[32][33];
    int b  = blockIdx.z;
    int c0 = blockIdx.x * 32;
    int r0 = blockIdx.y * 32;
    for (int i = threadIdx.y; i < 32; i += 8)
        tile[i][threadIdx.x] = in[((size_t)b * R + r0 + i) * Cc + c0 + threadIdx.x];
    __syncthreads();
    for (int i = threadIdx.y; i < 32; i += 8)
        out[((size_t)b * Cc + c0 + i) * R + r0 + threadIdx.x] = tile[threadIdx.x][i];
}

// h0[b,m,c] = dinv[m]^2 * xt[b,m,c]   (self-loop term of A_norm @ x)
__global__ void self_init_kernel(const float* __restrict__ xt, const float* __restrict__ dinv,
                                 float* __restrict__ h0, int Cc, int Np) {
    size_t i = (size_t)blockIdx.x * blockDim.x + threadIdx.x;
    int m = (int)((i / (unsigned)Cc) % (unsigned)Np);
    float d = dinv[m];
    h0[i] = d * d * xt[i];
}

// per (edge, batch) block: h0[b, s, :] += dinv[s]*dinv[t] * xt[b, t, :]
__global__ void scatter_kernel(const int* __restrict__ ei, int E,
                               const float* __restrict__ xt, const float* __restrict__ dinv,
                               float* __restrict__ h0, int Cc, int Np) {
    int idx = blockIdx.x;
    int e = idx % E;
    int b = idx / E;
    int s = ei[e];
    int t = ei[E + e];
    float wgt = dinv[s] * dinv[t];
    int c = threadIdx.x;
    atomicAdd(&h0[((size_t)b * Np + s) * Cc + c],
              wgt * xt[((size_t)b * Np + t) * Cc + c]);
}

__global__ void f32_to_f16_kernel(const float* __restrict__ in, _Float16* __restrict__ out, size_t n) {
    size_t i = (size_t)blockIdx.x * blockDim.x + threadIdx.x;
    if (i < n) out[i] = (_Float16)in[i];
}

// Wt[n*K + k] = (f16) W[k*N + n]   (column-major f16 weights for B-operand)
__global__ void wtrans_kernel(const float* __restrict__ W, _Float16* __restrict__ Wt, int K, int N) {
    int i = blockIdx.x * blockDim.x + threadIdx.x;
    if (i >= K * N) return;
    int k = i / N, n = i % N;
    Wt[(size_t)n * K + k] = (_Float16)W[i];
}

// ---------------- WMMA GEMM: C = A(f16,[M,K]) * Bt^T + bias (optional relu) ----------------
// 256 threads = 8 waves, block tile 64(M) x 256(N); wave tile 32 x 64:
// 8 wmma accumulators per wave, each B fragment reused by 2 WMMAs ->
// 12 b128 loads per 8 WMMAs per K-step (vs 10 per 4 previously).
template <bool RELU>
__global__ __launch_bounds__(256)
void gemm_wmma_f16(const _Float16* __restrict__ A, const _Float16* __restrict__ Bt,
                   const float* __restrict__ bias, float* __restrict__ Cmat,
                   int M, int N, int K) {
    const int lane   = threadIdx.x & 31;
    const int w      = threadIdx.x >> 5;
    const int wm     = w & 1;           // 2 waves along M
    const int wn     = w >> 1;          // 4 waves along N
    const int m_base = blockIdx.x * 64 + wm * 32;
    const int n_base = blockIdx.y * 256 + wn * 64;
    const int lr     = lane & 15;
    const int half   = lane >> 4;

    v8f acc[2][4];
#pragma unroll
    for (int t = 0; t < 2; ++t)
#pragma unroll
        for (int j = 0; j < 4; ++j)
#pragma unroll
            for (int i = 0; i < 8; ++i) acc[t][j][i] = 0.0f;

    // A layout (16-bit A 16x32): lane = (half,row); row = lr;
    // elems 0..15 = K[16h .. 16h+15] (contiguous, matches B)
    const _Float16* arow0 = A + (size_t)(m_base + lr) * K;
    const _Float16* arow1 = A + (size_t)(m_base + 16 + lr) * K;

    for (int kk = 0; kk < K; kk += 32) {
        v8h a0lo = *(const v8h*)(arow0 + kk + 16 * half);
        v8h a0hi = *(const v8h*)(arow0 + kk + 16 * half + 8);
        v8h a1lo = *(const v8h*)(arow1 + kk + 16 * half);
        v8h a1hi = *(const v8h*)(arow1 + kk + 16 * half + 8);
        v16h a0, a1;
#pragma unroll
        for (int i = 0; i < 8; ++i) {
            a0[i] = a0lo[i]; a0[i + 8] = a0hi[i];
            a1[i] = a1lo[i]; a1[i + 8] = a1hi[i];
        }

#pragma unroll
        for (int j = 0; j < 4; ++j) {
            // B layout (32x16 f16): lane lr = column n, half selects K 0..15 / 16..31,
            // 16 contiguous K values per lane -> contiguous in column-major Bt.
            const _Float16* bp = Bt + (size_t)(n_base + j * 16 + lr) * K + kk + 16 * half;
            v16h bv = *(const v16h*)bp;
            acc[0][j] = __builtin_amdgcn_wmma_f32_16x16x32_f16(
                false, a0, false, bv, (short)0, acc[0][j], false, false);
            acc[1][j] = __builtin_amdgcn_wmma_f32_16x16x32_f16(
                false, a1, false, bv, (short)0, acc[1][j], false, false);
        }
    }

    // D layout: VGPR r, m = r + 8*half, n = lr
#pragma unroll
    for (int t = 0; t < 2; ++t) {
#pragma unroll
        for (int j = 0; j < 4; ++j) {
            int n = n_base + j * 16 + lr;
            float bb = bias[n];
#pragma unroll
            for (int r = 0; r < 8; ++r) {
                int m = m_base + t * 16 + 8 * half + r;
                float v = acc[t][j][r] + bb;
                if (RELU) v = fmaxf(v, 0.0f);
                Cmat[(size_t)m * N + n] = v;
            }
        }
    }
}

// ---------------- row LayerNorm (one block per row), outputs f16 or f32 ----------------
template <typename T>
__global__ void ln_kernel(const float* __restrict__ in, const float* __restrict__ g,
                          const float* __restrict__ be, T* __restrict__ out, int D) {
    __shared__ float sh[256];
    __shared__ float sh2[256];
    int row = blockIdx.x;
    const float* p = in + (size_t)row * D;
    float s = 0.0f, s2 = 0.0f;
    for (int i = threadIdx.x; i < D; i += 256) { float v = p[i]; s += v; s2 += v * v; }
    sh[threadIdx.x] = s; sh2[threadIdx.x] = s2;
    __syncthreads();
    for (int st = 128; st > 0; st >>= 1) {
        if (threadIdx.x < st) { sh[threadIdx.x] += sh[threadIdx.x + st];
                                sh2[threadIdx.x] += sh2[threadIdx.x + st]; }
        __syncthreads();
    }
    float mean = sh[0] / D;
    float var  = sh2[0] / D - mean * mean;
    float rstd = rsqrtf(var + 1e-5f);
    T* q = out + (size_t)row * D;
    for (int i = threadIdx.x; i < D; i += 256) {
        float v = (p[i] - mean) * rstd * g[i] + be[i];
        q[i] = (T)v;
    }
}

// ---------------- host orchestration ----------------
extern "C" void kernel_launch(void* const* d_in, const int* in_sizes, int n_in,
                              void* d_out, int out_size, void* d_ws, size_t ws_size,
                              hipStream_t stream) {
    const float* x   = (const float*)d_in[0];
    const int*   ei  = (const int*)d_in[1];
    const float* W1  = (const float*)d_in[2];  const float* b1  = (const float*)d_in[3];
    const float* g1  = (const float*)d_in[4];  const float* be1 = (const float*)d_in[5];
    const float* W2  = (const float*)d_in[6];  const float* b2  = (const float*)d_in[7];
    const float* g2  = (const float*)d_in[8];  const float* be2 = (const float*)d_in[9];
    const float* W3  = (const float*)d_in[10]; const float* b3  = (const float*)d_in[11];
    const float* go  = (const float*)d_in[12]; const float* bo  = (const float*)d_in[13];

    const int B = 4, C = 256, Np = 64 * 64;          // N = 4096 nodes
    const int E = in_sizes[1] / 2;                   // 65536 edges
    const int M = B * Np;                            // 16384 rows
    const int N1 = 1024, N2 = 512, N3 = 256;

    char* ws = (char*)d_ws;
    size_t off = 0;
    auto alloc = [&](size_t bytes) -> void* {
        void* p = ws + off;
        off = (off + bytes + 255) & ~(size_t)255;
        return p;
    };
    int*      deg  = (int*)     alloc((size_t)Np * 4);
    float*    dinv = (float*)   alloc((size_t)Np * 4);
    float*    xt   = (float*)   alloc((size_t)M * C * 4);    // [B,N,C]; reused later for t3
    float*    h0   = (float*)   alloc((size_t)M * C * 4);    // reused later for t2h (f16, same bytes)
    _Float16* h16  = (_Float16*)alloc((size_t)M * C * 2);
    _Float16* W1t  = (_Float16*)alloc((size_t)N1 * C * 2);
    _Float16* W2t  = (_Float16*)alloc((size_t)N2 * N1 * 2);
    _Float16* W3t  = (_Float16*)alloc((size_t)N3 * N2 * 2);
    float*    t1   = (float*)   alloc((size_t)M * N1 * 4);   // reused later for t3n
    _Float16* t1h  = (_Float16*)alloc((size_t)M * N1 * 2);
    float*    t2   = (float*)   alloc((size_t)M * N2 * 4);
    _Float16* t2h  = (_Float16*)h0;                          // reuse (M*N2*2 <= M*C*4)
    float*    t3   = xt;                                      // reuse (M*N3*4 == M*C*4)
    float*    t3n  = t1;                                      // reuse

    // --- degree + normalization ---
    zero_i32_kernel<<<(Np + 255) / 256, 256, 0, stream>>>(deg, Np);
    degree_kernel<<<(E + 255) / 256, 256, 0, stream>>>(ei, E, deg);
    dinv_kernel<<<(Np + 255) / 256, 256, 0, stream>>>(deg, dinv, Np);

    // --- x [B,C,N] -> xt [B,N,C] ---
    transpose_bmn_kernel<<<dim3(Np / 32, C / 32, B), dim3(32, 8), 0, stream>>>(x, xt, C, Np);

    // --- sparse message passing: h0 = A_norm @ x ---
    self_init_kernel<<<(M * C) / 256, 256, 0, stream>>>(xt, dinv, h0, C, Np);
    scatter_kernel<<<E * B, C, 0, stream>>>(ei, E, xt, dinv, h0, C, Np);
    f32_to_f16_kernel<<<(M * C) / 256, 256, 0, stream>>>(h0, h16, (size_t)M * C);

    // --- weights -> f16 column-major ---
    wtrans_kernel<<<(C * N1 + 255) / 256, 256, 0, stream>>>(W1, W1t, C, N1);
    wtrans_kernel<<<(N1 * N2 + 255) / 256, 256, 0, stream>>>(W2, W2t, N1, N2);
    wtrans_kernel<<<(N2 * N3 + 255) / 256, 256, 0, stream>>>(W3, W3t, N2, N3);

    // --- MLP: gemm (+bias+relu) then LN (emits f16 input for next gemm) ---
    gemm_wmma_f16<true ><<<dim3(M / 64, N1 / 256), 256, 0, stream>>>(h16, W1t, b1, t1, M, N1, C);
    ln_kernel<_Float16><<<M, 256, 0, stream>>>(t1, g1, be1, t1h, N1);

    gemm_wmma_f16<true ><<<dim3(M / 64, N2 / 256), 256, 0, stream>>>(t1h, W2t, b2, t2, M, N2, N1);
    ln_kernel<_Float16><<<M, 256, 0, stream>>>(t2, g2, be2, t2h, N2);

    gemm_wmma_f16<false><<<dim3(M / 64, N3 / 256), 256, 0, stream>>>(t2h, W3t, b3, t3, M, N3, N2);
    ln_kernel<float><<<M, 256, 0, stream>>>(t3, go, bo, t3n, N3);

    // --- t3n [B,N,C] -> out [B,C,N] ---
    transpose_bmn_kernel<<<dim3(C / 32, Np / 32, B), dim3(32, 8), 0, stream>>>(
        t3n, (float*)d_out, Np, C);

    (void)n_in; (void)out_size; (void)ws_size;
}
